// RNNWavefunction_48928267436535
// MI455X (gfx1250) — matrix operand order repaired and checked by reading, not compiled, not run on hardware
//
#include <hip/hip_runtime.h>
#include <hip/hip_bf16.h>
#include <math.h>

typedef __bf16 bf16_t;
typedef __attribute__((ext_vector_type(16))) __bf16 v16bf;
typedef __attribute__((ext_vector_type(8)))  __bf16 v8bf;
typedef __attribute__((ext_vector_type(8)))  float  v8f;

#define H      128
#define SORB   64
#define WELEM  (3 * H * H)         // 49152 bf16 elements per weight matrix
#define LELEM  (4 * 512)           // 2048 bf16 elements for w_lin fragments
#define HSTAGE (32 * H)            // 4096 bf16 elements per 32-row h staging tile
// LDS: Wih1 + Whh1 fragments (192KB) + w_lin frags (4KB) + 4 waves * 2 * 8KB staging (64KB)
#define SWLIN_OFF  (2 * WELEM)
#define STAGE_OFF  (2 * WELEM + LELEM)
#define LDS_ELEMS  (STAGE_OFF + 8 * HSTAGE)
#define LDS_BYTES  (LDS_ELEMS * 2)          // 266240 B = 260 KiB (< 320 KiB cap)

__device__ __forceinline__ float fast_rcp(float x) { return __builtin_amdgcn_rcpf(x); }
__device__ __forceinline__ float fsigmoid(float x) { return fast_rcp(1.0f + __expf(-x)); }
__device__ __forceinline__ float ftanh(float x) {
#if __has_builtin(__builtin_amdgcn_tanhf)
    return __builtin_amdgcn_tanhf(x);            // native v_tanh_f32 on gfx1250
#else
    float e = __expf(2.0f * x);
    return (e - 1.0f) * fast_rcp(e + 1.0f);
#endif
}
__device__ __forceinline__ float lane_bcast(float v, int srcLane) {
    int iv = __builtin_bit_cast(int, v);
    int r  = __builtin_amdgcn_ds_bpermute(srcLane << 2, iv);   // LDS permute, no memory
    return __builtin_bit_cast(float, r);
}
__device__ __forceinline__ v8f vzero() { v8f z = {0,0,0,0,0,0,0,0}; return z; }
__device__ __forceinline__ v8f wmma1(v16bf a, v16bf b, v8f c) {
    return __builtin_amdgcn_wmma_f32_16x16x32_bf16(false, a, false, b, (short)0, c,
                                                   false, false);
}

// One gate tile column nt, 2x1 macro-tile: each B fragment (32B/lane, contiguous,
// pre-swizzled) is loaded once and feeds TWO WMMAs (M-tiles 0 and 1).
// wmat may point to LDS (layer 1) or global (layer 0) - addrspace inferred at inline site.
__device__ __forceinline__ void gemm2(const v16bf a0[4], const v16bf a1[4],
                                      const bf16_t* wmat, int nt, int lane,
                                      v8f& c0, v8f& c1) {
#pragma unroll
    for (int kt = 0; kt < 4; kt++) {
        v16bf b = *(const v16bf*)(wmat + (((nt << 2) | kt) << 9) + (lane << 4));
        c0 = wmma1(a0[kt], b, c0);
        c1 = wmma1(a1[kt], b, c1);
    }
}

// A fragments: lane holds row m=lane&15 of a 16-row tile; per K-tile of 32 the lane's
// K-run is [half*16,+16) -> two contiguous 16B ds_load_b128.
__device__ __forceinline__ void load_afrags(const bf16_t* sH, int m, int half, v16bf a[4]) {
#pragma unroll
    for (int kt = 0; kt < 4; kt++) {
        const v8bf* p0 = (const v8bf*)(sH + m * H + kt * 32 + half * 16);
        const v8bf* p1 = (const v8bf*)(sH + m * H + kt * 32 + half * 16 + 8);
        v8bf lo = *p0, hi = *p1;
        a[kt] = __builtin_shufflevector(lo, hi, 0,1,2,3,4,5,6,7,8,9,10,11,12,13,14,15);
    }
}

// ---- init kernel: Whh0 (384x128 f32) -> bf16, WMMA-B-fragment swizzled, in d_ws ----
// B[k][n] = W[n][k]; B frag layout: lane L holds col n=L&15, K = (L>>4)*16 + e.
__global__ void swizzle_whh0_kernel(const float* __restrict__ W, bf16_t* __restrict__ dst) {
    int pos = blockIdx.x * 256 + threadIdx.x;
    if (pos < WELEM) {
        int frag = pos >> 9, rem = pos & 511;
        int l = rem >> 4, e = rem & 15;
        int nt = frag >> 2, kt = frag & 3;
        int n = nt * 16 + (l & 15);
        int k = kt * 32 + (l >> 4) * 16 + e;
        dst[pos] = (bf16_t)W[n * H + k];
    }
}

__global__ void __launch_bounds__(128, 1)
rnnwf_kernel(const int* __restrict__ x,      const float* __restrict__ wih0,
             const bf16_t* __restrict__ gW0, const float* __restrict__ wih1,
             const float* __restrict__ whh1, const float* __restrict__ wlin,
             const float* __restrict__ blin, float* __restrict__ out, int B) {
    extern __shared__ __align__(32) unsigned char smem_raw[];
    bf16_t* sW = (bf16_t*)smem_raw;

    const int tid  = threadIdx.x;
    const int lane = tid & 31;
    const int wave = tid >> 5;
    const int half = lane >> 4;
    const int ln   = lane & 15;

    // ---- Phase 1: cooperative LDS fill (layer-1 weights + w_lin), bf16, B-swizzled ----
    const float* wsrc[2] = { wih1, whh1 };
#pragma unroll 1
    for (int mat = 0; mat < 2; mat++) {
        const float* W   = wsrc[mat];
        bf16_t*      dst = sW + mat * WELEM;
#pragma unroll 1
        for (int pos = tid; pos < WELEM; pos += 128) {
            int frag = pos >> 9, rem = pos & 511;
            int l = rem >> 4, e = rem & 15;
            int nt = frag >> 2, kt = frag & 3;
            int n = nt * 16 + (l & 15);
            int k = kt * 32 + (l >> 4) * 16 + e;
            dst[pos] = (bf16_t)W[n * H + k];
        }
    }
    // w_lin fragments (cols >= 2 zero-padded), nt fixed at 0 -> 4 K-tile fragments
    bf16_t* sWlin = sW + SWLIN_OFF;
#pragma unroll 1
    for (int pos = tid; pos < LELEM; pos += 128) {
        int kt = pos >> 9, rem = pos & 511;
        int l = rem >> 4, e = rem & 15;
        int n = l & 15;
        int k = kt * 32 + (l >> 4) * 16 + e;
        sWlin[pos] = (bf16_t)((n < 2) ? wlin[n * H + k] : 0.0f);
    }
    bf16_t* sH0 = sW + STAGE_OFF + wave * (2 * HSTAGE);
    bf16_t* sH1 = sH0 + HSTAGE;
    for (int p = lane; p < 2 * HSTAGE; p += 32) sH0[p] = (bf16_t)0.0f;
    __syncthreads();

    const float bl = (ln < 2) ? blin[ln] : 0.0f;
    const int   rowbase = blockIdx.x * 128 + wave * 32;   // 32 samples per wave
    const float PI = 3.14159265358979323846f;

    float amp = 1.0f, phase = 0.0f, num_up = 0.0f, num_down = 0.0f;
    unsigned prevmask = 0u;

#pragma unroll 1
    for (int i = 0; i < SORB; i++) {
        int xv = x[(rowbase + lane) * SORB + i];
#if __has_builtin(__builtin_amdgcn_ballot_w32)
        unsigned mask = (unsigned)__builtin_amdgcn_ballot_w32(xv > 0);
#else
        unsigned mask = (unsigned)__ballot(xv > 0);
#endif

        // ---------- layer 0: gh0 = h0_old @ Whh0^T (B-frags from L2-resident d_ws) ----
        v16bf A0[2][4];
        load_afrags(sH0,          ln, half, A0[0]);
        load_afrags(sH0 + 16 * H, ln, half, A0[1]);
#pragma unroll 1
        for (int jt = 0; jt < 8; jt++) {
            v8f aR[2] = { vzero(), vzero() };
            v8f aZ[2] = { vzero(), vzero() };
            v8f aN[2] = { vzero(), vzero() };
            gemm2(A0[0], A0[1], gW0, jt,      lane, aR[0], aR[1]);
            gemm2(A0[0], A0[1], gW0, jt + 8,  lane, aZ[0], aZ[1]);
            gemm2(A0[0], A0[1], gW0, jt + 16, lane, aN[0], aN[1]);
            int j = jt * 16 + ln;
            float gr0 = wih0[j * 2 + 0],           gr1 = wih0[j * 2 + 1];
            float gz0 = wih0[(H + j) * 2 + 0],     gz1 = wih0[(H + j) * 2 + 1];
            float gn0 = wih0[(2 * H + j) * 2 + 0], gn1 = wih0[(2 * H + j) * 2 + 1];
            float has = (i > 0) ? 1.0f : 0.0f;     // x0 starts at zeros
#pragma unroll
            for (int mt = 0; mt < 2; mt++) {
#pragma unroll
                for (int r = 0; r < 8; r++) {
                    int   m  = mt * 16 + r + 8 * half;
                    int   pb = (prevmask >> m) & 1;
                    float gir = has * (pb ? gr1 : gr0);
                    float giz = has * (pb ? gz1 : gz0);
                    float gin = has * (pb ? gn1 : gn0);
                    float hold = (float)sH0[m * H + j];
                    float rg = fsigmoid(gir + aR[mt][r]);
                    float zg = fsigmoid(giz + aZ[mt][r]);
                    float ng = ftanh(gin + rg * aN[mt][r]);
                    sH0[m * H + j] = (bf16_t)((1.0f - zg) * ng + zg * hold);
                }
            }
        }

        // ---------- layer 1: gi1 = h0_new @ Wih1^T, gh1 = h1_old @ Whh1^T (LDS) ------
        v16bf A0n[2][4], A1[2][4];
        load_afrags(sH0,          ln, half, A0n[0]);
        load_afrags(sH0 + 16 * H, ln, half, A0n[1]);
        load_afrags(sH1,          ln, half, A1[0]);
        load_afrags(sH1 + 16 * H, ln, half, A1[1]);
#pragma unroll 1
        for (int jt = 0; jt < 8; jt++) {
            v8f R[2]  = { vzero(), vzero() };
            v8f Z[2]  = { vzero(), vzero() };
            v8f GN[2] = { vzero(), vzero() };
            v8f HN[2] = { vzero(), vzero() };
            gemm2(A0n[0], A0n[1], sW,         jt,      lane, R[0],  R[1]);   // gi_r
            gemm2(A1[0],  A1[1],  sW + WELEM, jt,      lane, R[0],  R[1]);   // +gh_r
            gemm2(A0n[0], A0n[1], sW,         jt + 8,  lane, Z[0],  Z[1]);   // gi_z
            gemm2(A1[0],  A1[1],  sW + WELEM, jt + 8,  lane, Z[0],  Z[1]);   // +gh_z
            gemm2(A0n[0], A0n[1], sW,         jt + 16, lane, GN[0], GN[1]);  // gi_n
            gemm2(A1[0],  A1[1],  sW + WELEM, jt + 16, lane, HN[0], HN[1]);  // gh_n
            int j = jt * 16 + ln;
#pragma unroll
            for (int mt = 0; mt < 2; mt++) {
#pragma unroll
                for (int r = 0; r < 8; r++) {
                    int   m    = mt * 16 + r + 8 * half;
                    float hold = (float)sH1[m * H + j];
                    float rg = fsigmoid(R[mt][r]);
                    float zg = fsigmoid(Z[mt][r]);
                    float ng = ftanh(GN[mt][r] + rg * HN[mt][r]);
                    sH1[m * H + j] = (bf16_t)((1.0f - zg) * ng + zg * hold);
                }
            }
        }

        // ---------- linear head via WMMA + per-sample bookkeeping ----------
        v16bf A1n[2][4];
        load_afrags(sH1,          ln, half, A1n[0]);
        load_afrags(sH1 + 16 * H, ln, half, A1n[1]);
        v8f L[2] = { vzero(), vzero() };
#pragma unroll
        for (int kt = 0; kt < 4; kt++) {
            v16bf b = *(const v16bf*)(sWlin + (kt << 9) + (lane << 4));
            L[0] = wmma1(A1n[0][kt], b, L[0]);
            L[1] = wmma1(A1n[1][kt], b, L[1]);
        }
#pragma unroll
        for (int mt = 0; mt < 2; mt++)
#pragma unroll
            for (int r = 0; r < 8; r++) L[mt][r] += bl;

        // gather logits (C-layout: col in lane, row in VGPR) to sample-lane = row
        int mt_s = lane >> 4, mp = ln;
        int src0 = (mp >> 3) << 4, src1 = src0 + 1;
        float l0 = 0.0f, l1 = 0.0f;
#pragma unroll
        for (int mt = 0; mt < 2; mt++) {
#pragma unroll
            for (int r = 0; r < 8; r++) {
                float t0 = lane_bcast(L[mt][r], src0);
                float t1 = lane_bcast(L[mt][r], src1);
                if (mt_s == mt && (mp & 7) == r) { l0 = t0; l1 = t1; }
            }
        }

        int   bi = (mask >> lane) & 1;
        float mx = fmaxf(l0, l1);
        float e0 = __expf(l0 - mx), e1 = __expf(l1 - mx);
        float inv = fast_rcp(e0 + e1);
        float y0 = __builtin_sqrtf(e0 * inv), y1 = __builtin_sqrtf(e1 * inv);
        float p0 = PI * l0 * fast_rcp(1.0f + fabsf(l0));
        float p1 = PI * l1 * fast_rcp(1.0f + fabsf(l1));
        bool  is_even = (i & 1) == 0;
        float num   = is_even ? num_up : num_down;
        float lower = (float)(-16 + (i >> 1));            // baseline + i//2
        float occ   = (num < 16.0f) ? 1.0f : 0.0f;        // alpha = 16
        float unocc = (num > lower) ? 1.0f : 0.0f;
        float m0 = y0 * unocc, m1 = y1 * occ;
        float nrm = fmaxf(__builtin_sqrtf(m0 * m0 + m1 * m1), 1e-12f);
        if (i >= 16) { float rn = fast_rcp(nrm); y0 = m0 * rn; y1 = m1 * rn; }
        float bf = (float)bi;
        if (is_even) num_up += bf; else num_down += bf;
        amp   *= bi ? y1 : y0;
        phase += bi ? p1 : p0;
        prevmask = mask;
    }

    {
        int gm = rowbase + lane;
        out[gm]     = amp * __cosf(phase);
        out[B + gm] = amp * __sinf(phase);
    }
}

extern "C" void kernel_launch(void* const* d_in, const int* in_sizes, int n_in,
                              void* d_out, int out_size, void* d_ws, size_t ws_size,
                              hipStream_t stream) {
    (void)n_in; (void)ws_size; (void)out_size;
    const int*   x    = (const int*)d_in[0];
    const float* wih0 = (const float*)d_in[1];
    const float* whh0 = (const float*)d_in[2];
    const float* wih1 = (const float*)d_in[3];
    const float* whh1 = (const float*)d_in[4];
    const float* wlin = (const float*)d_in[5];
    const float* blin = (const float*)d_in[6];
    float*       out  = (float*)d_out;
    bf16_t*      gW0  = (bf16_t*)d_ws;      // 96KB pre-swizzled Whh0, L2-resident

    int B    = in_sizes[0] / SORB;   // 65536
    int grid = B / 128;              // 128 samples per workgroup (4 waves x 32)

    swizzle_whh0_kernel<<<(WELEM + 255) / 256, 256, 0, stream>>>(whh0, gW0);

    hipFuncSetAttribute((const void*)rnnwf_kernel,
                        hipFuncAttributeMaxDynamicSharedMemorySize, LDS_BYTES);
    rnnwf_kernel<<<grid, 128, LDS_BYTES, stream>>>(x, wih0, gW0, wih1, whh1,
                                                   wlin, blin, out, B);
}
